// DLRM_Net_68719476736559
// MI455X (gfx1250) — matrix operand order repaired and checked by reading, not compiled, not run on hardware
//
#include <hip/hip_runtime.h>
#include <hip/hip_bf16.h>
#include <math.h>

typedef __attribute__((ext_vector_type(16))) _Float16 v16h;
typedef __attribute__((ext_vector_type(8)))  _Float16 v8h;
typedef __attribute__((ext_vector_type(4)))  _Float16 v4h;
typedef __attribute__((ext_vector_type(8)))  float    v8f;

#define BATCH      4096
#define NUM_TABLES 26
#define EMB_DIM    128
#define VOCAB      100000
#define POOL_L     4
#define NFEAT      27      // 1 dense row + 26 pooled rows
#define NPAIRS     351     // strict lower triangle of 27x27
#define RDIM       479
#define RPITCH     480     // 15*32, padded K for top-MLP layer 0

// ---------------------------------------------------------------------------
// Staging: f32 (rows x cols) -> f16 (rowsPad x colsPad), zero padding.
// ---------------------------------------------------------------------------
__global__ void pad_f32_to_f16(const float* __restrict__ src,
                               _Float16* __restrict__ dst,
                               int rows, int cols, int rowsPad, int colsPad)
{
    int idx = blockIdx.x * blockDim.x + threadIdx.x;
    int total = rowsPad * colsPad;
    if (idx >= total) return;
    int r = idx / colsPad;
    int c = idx - r * colsPad;
    float v = (r < rows && c < cols) ? src[(size_t)r * cols + c] : 0.0f;
    dst[idx] = (_Float16)v;
}

__global__ void pad_bias(const float* __restrict__ src,
                         float* __restrict__ dst, int n, int npad)
{
    int i = blockIdx.x * blockDim.x + threadIdx.x;
    if (i < npad) dst[i] = (i < n) ? src[i] : 0.0f;
}

__global__ void zero_f16(_Float16* __restrict__ p, int n)
{
    int i = blockIdx.x * blockDim.x + threadIdx.x;
    if (i < n) p[i] = (_Float16)0.0f;
}

// ---------------------------------------------------------------------------
// WMMA GEMM: C = act(A[M,Kpad] @ W[Npad,Kpad]^T + bias[Npad])
// Pre-padded f16 operands, branch-free b128 inner loop, 8 wmma / k-step.
// One wave -> 64x32 output block (4 M-tiles x 2 N-tiles).
// mode 0: relu -> f16 C16 (pitch Npad); mode 1: sigmoid -> f32 C32.
// ---------------------------------------------------------------------------
__global__ __launch_bounds__(32)
void wmma_gemm(const _Float16* __restrict__ A,
               const _Float16* __restrict__ W,
               const float* __restrict__ bias,
               _Float16* __restrict__ C16,
               float* __restrict__ C32,
               int Kpad, int Npad, int NReal, int mode)
{
    const int lane = threadIdx.x & 31;
    const int mrow = lane & 15;
    const int kh   = lane >> 4;
    const int mBase = blockIdx.y * 64;
    const int nBase = blockIdx.x * 32;

    const _Float16* Ar0 = A + (size_t)(mBase +  0 + mrow) * Kpad;
    const _Float16* Ar1 = A + (size_t)(mBase + 16 + mrow) * Kpad;
    const _Float16* Ar2 = A + (size_t)(mBase + 32 + mrow) * Kpad;
    const _Float16* Ar3 = A + (size_t)(mBase + 48 + mrow) * Kpad;
    const _Float16* Wr0 = W + (size_t)(nBase +  0 + (lane & 15)) * Kpad;
    const _Float16* Wr1 = W + (size_t)(nBase + 16 + (lane & 15)) * Kpad;

    v8f acc[4][2] = {};

    for (int k0 = 0; k0 < Kpad; k0 += 32) {
        v8h bl0 = *(const v8h*)(Wr0 + k0 + kh * 16);
        v8h bh0 = *(const v8h*)(Wr0 + k0 + kh * 16 + 8);
        v8h bl1 = *(const v8h*)(Wr1 + k0 + kh * 16);
        v8h bh1 = *(const v8h*)(Wr1 + k0 + kh * 16 + 8);
        v16h bf0 = __builtin_shufflevector(bl0, bh0, 0,1,2,3,4,5,6,7,8,9,10,11,12,13,14,15);
        v16h bf1 = __builtin_shufflevector(bl1, bh1, 0,1,2,3,4,5,6,7,8,9,10,11,12,13,14,15);

        const _Float16* Ars[4] = {Ar0, Ar1, Ar2, Ar3};
#pragma unroll
        for (int mi = 0; mi < 4; ++mi) {
            v8h al = *(const v8h*)(Ars[mi] + k0 + kh * 16);
            v8h ah = *(const v8h*)(Ars[mi] + k0 + kh * 16 + 8);
            v16h af = __builtin_shufflevector(al, ah, 0,1,2,3,4,5,6,7,8,9,10,11,12,13,14,15);
            acc[mi][0] = __builtin_amdgcn_wmma_f32_16x16x32_f16(
                false, af, false, bf0, (short)0, acc[mi][0], false, false);
            acc[mi][1] = __builtin_amdgcn_wmma_f32_16x16x32_f16(
                false, af, false, bf1, (short)0, acc[mi][1], false, false);
        }
    }

#pragma unroll
    for (int nj = 0; nj < 2; ++nj) {
        const int nout = nBase + nj * 16 + (lane & 15);
        const float bv = bias[nout];
#pragma unroll
        for (int mi = 0; mi < 4; ++mi) {
#pragma unroll
            for (int r = 0; r < 8; ++r) {
                int mo = mBase + mi * 16 + r + 8 * kh;
                float v = acc[mi][nj][r] + bv;
                if (mode == 0) {
                    v = v > 0.0f ? v : 0.0f;
                    C16[(size_t)mo * Npad + nout] = (_Float16)v;
                } else if (nout < NReal) {
                    v = 1.0f / (1.0f + __expf(-v));
                    C32[(size_t)mo * NReal + nout] = v;
                }
            }
        }
    }
}

// ---------------------------------------------------------------------------
// Embedding sum-pooling -> f16 pooled [table][batch][EMB_DIM].
// One wave per (table, sample); lane owns 4 columns (b128 gathers).
// Fast path issues all 4 row gathers concurrently (latency-bound kernel).
// ---------------------------------------------------------------------------
__global__ __launch_bounds__(32)
void pool_kernel(const float* __restrict__ tables,
                 const int*   __restrict__ indices,
                 const int*   __restrict__ offsets,
                 _Float16* __restrict__ pooled)
{
    const int blk  = blockIdx.x;          // t * BATCH + b
    const int t    = blk / BATCH;
    const int b    = blk - t * BATCH;
    const int lane = threadIdx.x;

    const int start = offsets[(size_t)t * BATCH + b];
    const int end   = (b + 1 < BATCH) ? offsets[(size_t)t * BATCH + b + 1]
                                      : BATCH * POOL_L;

    const float* tab  = tables + (size_t)t * VOCAB * EMB_DIM;
    const int*   idxp = indices + (size_t)t * BATCH * POOL_L;

    float4 s = make_float4(0.f, 0.f, 0.f, 0.f);
    if (end - start == POOL_L) {
        int i0 = idxp[start + 0], i1 = idxp[start + 1];
        int i2 = idxp[start + 2], i3 = idxp[start + 3];
        float4 v0 = ((const float4*)(tab + (size_t)i0 * EMB_DIM))[lane];
        float4 v1 = ((const float4*)(tab + (size_t)i1 * EMB_DIM))[lane];
        float4 v2 = ((const float4*)(tab + (size_t)i2 * EMB_DIM))[lane];
        float4 v3 = ((const float4*)(tab + (size_t)i3 * EMB_DIM))[lane];
        s.x = (v0.x + v1.x) + (v2.x + v3.x);
        s.y = (v0.y + v1.y) + (v2.y + v3.y);
        s.z = (v0.z + v1.z) + (v2.z + v3.z);
        s.w = (v0.w + v1.w) + (v2.w + v3.w);
    } else {
        for (int r = start; r < end; ++r) {
            int idx = idxp[r];
            float4 v = ((const float4*)(tab + (size_t)idx * EMB_DIM))[lane];
            s.x += v.x; s.y += v.y; s.z += v.z; s.w += v.w;
        }
    }
    v4h o; o[0] = (_Float16)s.x; o[1] = (_Float16)s.y;
           o[2] = (_Float16)s.z; o[3] = (_Float16)s.w;
    *(v4h*)(pooled + ((size_t)t * BATCH + b) * EMB_DIM + lane * 4) = o;
}

// ---------------------------------------------------------------------------
// WMMA interaction: per sample, Z = T T^T with T (27x128 f16, padded to 32
// rows via zero-row pointer). Both A and B fragments are contiguous K-runs
// from the same row-major T rows (B cols of T^T == rows of T), so fragment
// loads are plain b128s. 4 k-steps x 4 wmma per sample; scatter strict lower
// triangle into R f16 (pitch 480).
// ---------------------------------------------------------------------------
__device__ __forceinline__ const _Float16*
trow(int r, int b, const _Float16* x128, const _Float16* pooled,
     const _Float16* zrow)
{
    if (r == 0)     return x128 + (size_t)b * EMB_DIM;
    if (r < NFEAT)  return pooled + ((size_t)(r - 1) * BATCH + b) * EMB_DIM;
    return zrow;
}

__global__ __launch_bounds__(32)
void interact_wmma(const _Float16* __restrict__ x128,
                   const _Float16* __restrict__ pooled,
                   const _Float16* __restrict__ zrow,
                   _Float16* __restrict__ R)
{
    const int b    = blockIdx.x;
    const int lane = threadIdx.x & 31;
    const int lr   = lane & 15;
    const int kh   = lane >> 4;

    // per-lane row pointers: tile 0 = rows 0..15, tile 1 = rows 16..31
    const _Float16* row0 = trow(lr,      b, x128, pooled, zrow);
    const _Float16* row1 = trow(16 + lr, b, x128, pooled, zrow);
    const _Float16* rows[2] = {row0, row1};

    v8f acc[2][2] = {};
#pragma unroll
    for (int k0 = 0; k0 < EMB_DIM; k0 += 32) {
        v16h af[2], bf[2];
#pragma unroll
        for (int tile = 0; tile < 2; ++tile) {
            v8h bl = *(const v8h*)(rows[tile] + k0 + kh * 16);
            v8h bh = *(const v8h*)(rows[tile] + k0 + kh * 16 + 8);
            bf[tile] = __builtin_shufflevector(bl, bh, 0,1,2,3,4,5,6,7,8,9,10,11,12,13,14,15);
            v8h al = *(const v8h*)(rows[tile] + k0 + kh * 16);
            v8h ah = *(const v8h*)(rows[tile] + k0 + kh * 16 + 8);
            af[tile] = __builtin_shufflevector(al, ah, 0,1,2,3,4,5,6,7,8,9,10,11,12,13,14,15);
        }
#pragma unroll
        for (int mi = 0; mi < 2; ++mi)
#pragma unroll
            for (int nj = 0; nj < 2; ++nj)
                acc[mi][nj] = __builtin_amdgcn_wmma_f32_16x16x32_f16(
                    false, af[mi], false, bf[nj], (short)0, acc[mi][nj], false, false);
    }

    _Float16* Rb = R + (size_t)b * RPITCH;
    // dense-x copy + pad column
    const _Float16* xb = x128 + (size_t)b * EMB_DIM;
    for (int c = lane; c < EMB_DIM; c += 32) Rb[c] = xb[c];
    if (lane == 0) Rb[RDIM] = (_Float16)0.0f;

    // scatter Z strict lower triangle: p = i*(i-1)/2 + j
#pragma unroll
    for (int mi = 0; mi < 2; ++mi)
#pragma unroll
        for (int nj = 0; nj < 2; ++nj)
#pragma unroll
            for (int r = 0; r < 8; ++r) {
                int i = mi * 16 + r + 8 * kh;
                int j = nj * 16 + lr;
                if (i < NFEAT && j < i)
                    Rb[EMB_DIM + (i * (i - 1)) / 2 + j] = (_Float16)acc[mi][nj][r];
            }
}

// ---------------------------------------------------------------------------
// Host side
// ---------------------------------------------------------------------------
static inline size_t alignUp(size_t x, size_t a) { return (x + a - 1) & ~(a - 1); }

extern "C" void kernel_launch(void* const* d_in, const int* in_sizes, int n_in,
                              void* d_out, int out_size, void* d_ws, size_t ws_size,
                              hipStream_t stream)
{
    (void)in_sizes; (void)n_in; (void)out_size; (void)ws_size;

    const float* dense_x = (const float*)d_in[0];
    const float* emb     = (const float*)d_in[1];
    const float* Wsrc[8] = {(const float*)d_in[2],  (const float*)d_in[4],
                            (const float*)d_in[6],  (const float*)d_in[8],
                            (const float*)d_in[10], (const float*)d_in[12],
                            (const float*)d_in[14], (const float*)d_in[16]};
    const float* Bsrc[8] = {(const float*)d_in[3],  (const float*)d_in[5],
                            (const float*)d_in[7],  (const float*)d_in[9],
                            (const float*)d_in[11], (const float*)d_in[13],
                            (const float*)d_in[15], (const float*)d_in[17]};
    const int*   idx = (const int*)d_in[18];
    const int*   off = (const int*)d_in[19];

    const int Kr[8] = {13, 512, 256, RDIM, 1024, 1024, 512, 256};
    const int Nr[8] = {512, 256, 128, 1024, 1024, 512, 256, 1};
    const int Kp[8] = {32, 512, 256, RPITCH, 1024, 1024, 512, 256};
    const int Np[8] = {512, 256, 128, 1024, 1024, 512, 256, 32};

    char* p = (char*)d_ws;
    auto take = [&](size_t bytes) { char* r = p; p += alignUp(bytes, 256); return r; };

    _Float16* pooled = (_Float16*)take((size_t)NUM_TABLES * BATCH * EMB_DIM * 2);
    _Float16* bufA   = (_Float16*)take((size_t)BATCH * 1024 * 2);
    _Float16* bufB   = (_Float16*)take((size_t)BATCH * 1024 * 2);
    _Float16* x128h  = (_Float16*)take((size_t)BATCH * EMB_DIM * 2);
    _Float16* A0     = (_Float16*)take((size_t)BATCH * 32 * 2);
    _Float16* zrow   = (_Float16*)take((size_t)EMB_DIM * 2);
    _Float16* Wh[8];
    float*    Bh[8];
    for (int l = 0; l < 8; ++l) Wh[l] = (_Float16*)take((size_t)Np[l] * Kp[l] * 2);
    for (int l = 0; l < 8; ++l) Bh[l] = (float*)take((size_t)Np[l] * 4);

    // ---- staging ----
    for (int l = 0; l < 8; ++l) {
        int total = Np[l] * Kp[l];
        pad_f32_to_f16<<<(total + 255) / 256, 256, 0, stream>>>(
            Wsrc[l], Wh[l], Nr[l], Kr[l], Np[l], Kp[l]);
        pad_bias<<<(Np[l] + 255) / 256, 256, 0, stream>>>(Bsrc[l], Bh[l], Nr[l], Np[l]);
    }
    pad_f32_to_f16<<<(BATCH * 32 + 255) / 256, 256, 0, stream>>>(
        dense_x, A0, BATCH, 13, BATCH, 32);
    zero_f16<<<1, 128, 0, stream>>>(zrow, EMB_DIM);

    const dim3 wave(32);
    const int  mb = BATCH / 64;

    // ---- bottom MLP: 13->512->256->128 ----
    wmma_gemm<<<dim3(Np[0] / 32, mb), wave, 0, stream>>>(A0,   Wh[0], Bh[0], bufA,  nullptr, Kp[0], Np[0], Nr[0], 0);
    wmma_gemm<<<dim3(Np[1] / 32, mb), wave, 0, stream>>>(bufA, Wh[1], Bh[1], bufB,  nullptr, Kp[1], Np[1], Nr[1], 0);
    wmma_gemm<<<dim3(Np[2] / 32, mb), wave, 0, stream>>>(bufB, Wh[2], Bh[2], x128h, nullptr, Kp[2], Np[2], Nr[2], 0);

    // ---- embedding pooling (HBM roofline term) ----
    pool_kernel<<<NUM_TABLES * BATCH, wave, 0, stream>>>(emb, idx, off, pooled);

    // ---- WMMA interaction -> R (f16, pitch 480) in bufA ----
    interact_wmma<<<BATCH, wave, 0, stream>>>(x128h, pooled, zrow, bufA);

    // ---- top MLP: 479->1024->1024->512->256->1 ----
    wmma_gemm<<<dim3(Np[3] / 32, mb), wave, 0, stream>>>(bufA, Wh[3], Bh[3], bufB, nullptr, Kp[3], Np[3], Nr[3], 0);
    wmma_gemm<<<dim3(Np[4] / 32, mb), wave, 0, stream>>>(bufB, Wh[4], Bh[4], bufA, nullptr, Kp[4], Np[4], Nr[4], 0);
    wmma_gemm<<<dim3(Np[5] / 32, mb), wave, 0, stream>>>(bufA, Wh[5], Bh[5], bufB, nullptr, Kp[5], Np[5], Nr[5], 0);
    wmma_gemm<<<dim3(Np[6] / 32, mb), wave, 0, stream>>>(bufB, Wh[6], Bh[6], bufA, nullptr, Kp[6], Np[6], Nr[6], 0);
    wmma_gemm<<<dim3(Np[7] / 32, mb), wave, 0, stream>>>(bufA, Wh[7], Bh[7], nullptr, (float*)d_out, Kp[7], Np[7], Nr[7], 1);
}
